// Attention_18064632447308
// MI455X (gfx1250) — matrix operand order repaired and checked by reading, not compiled, not run on hardware
//
#include <hip/hip_runtime.h>

typedef __attribute__((ext_vector_type(16))) _Float16 v16h;
typedef __attribute__((ext_vector_type(8)))  float    v8f;

#define LSEQ 2048   // n*h
#define DMODEL 256
#define NBATCH 16
#define NE 768      // 3*d

union FragU { uint4 u[2]; v16h h; };

__device__ __forceinline__ float fast_exp2(float x) { return __builtin_amdgcn_exp2f(x); }

// ---------------------------------------------------------------------------
// Kernel 1: QKV projection  lin = x @ W^T, de-interleaved into Q, K, Vt (f16)
//   Q pre-scaled by d^-0.5 * log2(e)  (so attention uses exp2 directly)
//   Vt stored transposed [256][L] per batch for contiguous B-fragment loads.
// grid = 2048 (M-tiles of 16 tokens over b*L = 32768), block = 128 (4 waves)
// ---------------------------------------------------------------------------
__global__ __launch_bounds__(128) void qkv_proj_kernel(
    const float* __restrict__ x, const float* __restrict__ W,
    _Float16* __restrict__ qb, _Float16* __restrict__ kb,
    _Float16* __restrict__ vtb)
{
  const int lane = threadIdx.x & 31;
  const int wave = threadIdx.x >> 5;
  const int l16  = lane & 15;
  const int g    = lane >> 4;
  const int mtile = blockIdx.x;

  // A fragments (16x32 f16): lane holds row M=l16; K = c*32 + (h/8)*16 + h%8 + g*8
  const int t = mtile * 16 + l16;
  const float* xrow = x + (size_t)t * DMODEL;
  v16h A[8];
#pragma unroll
  for (int c = 0; c < 8; ++c) {
    const float4* p0 = (const float4*)(xrow + c * 32 + g * 8);
    const float4* p1 = (const float4*)(xrow + c * 32 + 16 + g * 8);
    float4 a0 = p0[0], a1 = p0[1], b0 = p1[0], b1 = p1[1];
    v16h r;
    r[0]=(_Float16)a0.x; r[1]=(_Float16)a0.y; r[2]=(_Float16)a0.z; r[3]=(_Float16)a0.w;
    r[4]=(_Float16)a1.x; r[5]=(_Float16)a1.y; r[6]=(_Float16)a1.z; r[7]=(_Float16)a1.w;
    r[8]=(_Float16)b0.x; r[9]=(_Float16)b0.y; r[10]=(_Float16)b0.z; r[11]=(_Float16)b0.w;
    r[12]=(_Float16)b1.x; r[13]=(_Float16)b1.y; r[14]=(_Float16)b1.z; r[15]=(_Float16)b1.w;
    A[c] = r;
  }

  const float qscale = 0.0625f * 1.44269504088896340736f;  // 256^-0.5 * log2(e)

#pragma unroll 1
  for (int j = 0; j < 12; ++j) {
    const int nt = wave * 12 + j;            // 48 N-tiles over 768 features
    const int e  = nt * 16 + l16;            // this lane's output column
    const float* wrow = W + (size_t)e * DMODEL;
    v8f C = {};
#pragma unroll
    for (int c = 0; c < 8; ++c) {
      // B fragment (32x16): lane holds col N=l16; K = c*32 + (h/8)*16 + h%8 + g*8
      const float4* p = (const float4*)(wrow + c * 32 + g * 8);
      const float4* q = (const float4*)(wrow + c * 32 + 16 + g * 8);
      float4 w0 = p[0], w1 = p[1], w2 = q[0], w3 = q[1];
      v16h Bf;
      Bf[0]=(_Float16)w0.x; Bf[1]=(_Float16)w0.y; Bf[2]=(_Float16)w0.z; Bf[3]=(_Float16)w0.w;
      Bf[4]=(_Float16)w1.x; Bf[5]=(_Float16)w1.y; Bf[6]=(_Float16)w1.z; Bf[7]=(_Float16)w1.w;
      Bf[8]=(_Float16)w2.x; Bf[9]=(_Float16)w2.y; Bf[10]=(_Float16)w2.z; Bf[11]=(_Float16)w2.w;
      Bf[12]=(_Float16)w3.x; Bf[13]=(_Float16)w3.y; Bf[14]=(_Float16)w3.z; Bf[15]=(_Float16)w3.w;
      C = __builtin_amdgcn_wmma_f32_16x16x32_f16(false, A[c], false, Bf,
                                                 (short)0, C, false, false);
    }
    // de-interleave store: e -> (dd = e/3, which = e%3)
    const int dd = e / 3;
    const int which = e - dd * 3;
#pragma unroll
    for (int r = 0; r < 8; ++r) {
      const int tt = mtile * 16 + r + g * 8;  // C row M = r + 8*(lane/16)
      const float v = C[r];
      if (which == 0) {
        qb[(size_t)tt * DMODEL + dd] = (_Float16)(v * qscale);
      } else if (which == 1) {
        kb[(size_t)tt * DMODEL + dd] = (_Float16)v;
      } else {
        const int bidx = tt >> 11, ii = tt & (LSEQ - 1);
        vtb[((size_t)bidx * DMODEL + dd) * LSEQ + ii] = (_Float16)v;
      }
    }
  }
}

// ---------------------------------------------------------------------------
// Kernel 2: flash attention. Each wave owns 16 query rows, loops 64 key-tiles
// of 32. Q staged in padded LDS; P transposed C->A layout through padded LDS.
// Next key-tile's K/V lines prefetched (global_prefetch_b8) under current
// tile's WMMAs to cover L2->WGP$ latency at low occupancy.
// grid = 16 batches * 32 blocks, block = 128 (4 waves => 64 query rows/block)
// ---------------------------------------------------------------------------
__global__ __launch_bounds__(128) void flash_attn_kernel(
    const _Float16* __restrict__ qb, const _Float16* __restrict__ kb,
    const _Float16* __restrict__ vtb, float* __restrict__ out)
{
  __shared__ __align__(16) _Float16 qlds[4][16][264];  // 264 = 256 + 8 pad (bank spread)
  __shared__ __align__(16) _Float16 plds[4][16][40];   // 40  = 32  + 8 pad

  const int lane = threadIdx.x & 31;
  const int wave = threadIdx.x >> 5;
  const int l16  = lane & 15;
  const int g    = lane >> 4;

  const int b  = blockIdx.x >> 5;                    // 32 blocks per batch
  const int i0 = (blockIdx.x & 31) * 64 + wave * 16; // query row within batch

  const _Float16* Qrow = qb  + ((size_t)b * LSEQ + i0) * DMODEL;
  const _Float16* Kb   = kb  + (size_t)b * LSEQ * DMODEL;
  const _Float16* Vtb  = vtb + (size_t)b * DMODEL * LSEQ;

  // Stage this wave's 16x256 Q tile into padded LDS (lane: row l16, half g)
  {
    const _Float16* src = Qrow + (size_t)l16 * DMODEL + g * 128;
#pragma unroll
    for (int jj = 0; jj < 16; ++jj) {
      uint4 v = *(const uint4*)(src + jj * 8);
      *(uint4*)&qlds[wave][l16][g * 128 + jj * 8] = v;
    }
  }

  v8f O[16];
#pragma unroll
  for (int n = 0; n < 16; ++n) O[n] = {};
  float m[8], lsum[8];
#pragma unroll
  for (int r = 0; r < 8; ++r) { m[r] = -3.0e38f; lsum[r] = 0.0f; }

#pragma unroll 1
  for (int kt = 0; kt < 64; ++kt) {
    const int kbase = kt * 32;

    // ---- prefetch next key tile (K: 32KB contiguous; V: 256 x 64B strided) ----
    if (kt < 63) {
      const int nbase = kbase + 32;
      // K rows [nbase, nbase+64) are contiguous: 32KB = 256 x 128B lines
      const char* kpf = (const char*)(Kb + (size_t)nbase * DMODEL);
#pragma unroll
      for (int p = 0; p < 8; ++p)
        __builtin_prefetch(kpf + (size_t)(p * 32 + lane) * 128, 0, 3);
      // V: one 64B segment per d-row at stride 4096B; 256 rows via 8 waves of lanes
      const char* vpf = (const char*)(Vtb + nbase);
#pragma unroll
      for (int p = 0; p < 8; ++p)
        __builtin_prefetch(vpf + (size_t)(p * 32 + lane) * (LSEQ * 2), 0, 3);
    }

    // ---- S = Qscaled @ K^T for 16 queries x 32 keys ----
    v8f S0 = {}, S1 = {};
#pragma unroll
    for (int c = 0; c < 8; ++c) {
      FragU fa;
      fa.u[0] = *(const uint4*)&qlds[wave][l16][c * 32 + g * 8];
      fa.u[1] = *(const uint4*)&qlds[wave][l16][c * 32 + 16 + g * 8];
      const _Float16* k0 = Kb + (size_t)(kbase + l16)      * DMODEL + c * 32 + g * 8;
      const _Float16* k1 = Kb + (size_t)(kbase + 16 + l16) * DMODEL + c * 32 + g * 8;
      FragU fb0, fb1;
      fb0.u[0] = *(const uint4*)k0; fb0.u[1] = *(const uint4*)(k0 + 16);
      fb1.u[0] = *(const uint4*)k1; fb1.u[1] = *(const uint4*)(k1 + 16);
      S0 = __builtin_amdgcn_wmma_f32_16x16x32_f16(false, fa.h, false, fb0.h,
                                                  (short)0, S0, false, false);
      S1 = __builtin_amdgcn_wmma_f32_16x16x32_f16(false, fa.h, false, fb1.h,
                                                  (short)0, S1, false, false);
    }

    // ---- online softmax (rows M = r + 8g live across lanes with same g) ----
    float alpha[8];
#pragma unroll
    for (int r = 0; r < 8; ++r) {
      float s0 = S0[r], s1 = S1[r];
      float rm = fmaxf(s0, s1);
      rm = fmaxf(rm, __shfl_xor(rm, 1, 32));
      rm = fmaxf(rm, __shfl_xor(rm, 2, 32));
      rm = fmaxf(rm, __shfl_xor(rm, 4, 32));
      rm = fmaxf(rm, __shfl_xor(rm, 8, 32));
      const float mnew = fmaxf(m[r], rm);
      const float a  = fast_exp2(m[r] - mnew);
      const float p0 = fast_exp2(s0 - mnew);
      const float p1 = fast_exp2(s1 - mnew);
      float rs = p0 + p1;
      rs += __shfl_xor(rs, 1, 32);
      rs += __shfl_xor(rs, 2, 32);
      rs += __shfl_xor(rs, 4, 32);
      rs += __shfl_xor(rs, 8, 32);
      lsum[r] = lsum[r] * a + rs;
      m[r] = mnew;
      alpha[r] = a;
      // stash P (f16) in C-layout position for the LDS transpose
      plds[wave][r + g * 8][l16]      = (_Float16)p0;
      plds[wave][r + g * 8][16 + l16] = (_Float16)p1;
    }

    // rescale accumulated O by alpha (per row)
#pragma unroll
    for (int n = 0; n < 16; ++n)
#pragma unroll
      for (int r = 0; r < 8; ++r) O[n][r] *= alpha[r];

    // reload P in A-fragment layout (same-wave LDS ops are in order)
    FragU fp;
    fp.u[0] = *(const uint4*)&plds[wave][l16][g * 8];
    fp.u[1] = *(const uint4*)&plds[wave][l16][16 + g * 8];

    // ---- O += P @ V over d = 256 (16 N-tiles), Vt contiguous in keys ----
#pragma unroll
    for (int n = 0; n < 16; ++n) {
      const _Float16* vp = Vtb + (size_t)(n * 16 + l16) * LSEQ + kbase + g * 8;
      FragU fv;
      fv.u[0] = *(const uint4*)vp; fv.u[1] = *(const uint4*)(vp + 16);
      O[n] = __builtin_amdgcn_wmma_f32_16x16x32_f16(false, fp.h, false, fv.h,
                                                    (short)0, O[n], false, false);
    }
  }

  // ---- normalize and write out [b][i][d] (coalesced over l16) ----
#pragma unroll
  for (int r = 0; r < 8; ++r) lsum[r] = 1.0f / lsum[r];
  float* orow = out + ((size_t)b * LSEQ + i0) * DMODEL;
#pragma unroll
  for (int n = 0; n < 16; ++n)
#pragma unroll
    for (int r = 0; r < 8; ++r)
      orow[(size_t)(r + g * 8) * DMODEL + n * 16 + l16] = O[n][r] * lsum[r];
}

// ---------------------------------------------------------------------------
extern "C" void kernel_launch(void* const* d_in, const int* in_sizes, int n_in,
                              void* d_out, int out_size, void* d_ws, size_t ws_size,
                              hipStream_t stream) {
  (void)in_sizes; (void)n_in; (void)out_size; (void)ws_size;
  const float* x = (const float*)d_in[0];
  const float* W = (const float*)d_in[1];
  float* out = (float*)d_out;

  const size_t tok = (size_t)NBATCH * LSEQ * DMODEL;  // 8.4M f16 elements each
  _Float16* qb  = (_Float16*)d_ws;
  _Float16* kb  = qb + tok;
  _Float16* vtb = kb + tok;

  // 2048 M-tiles of 16 tokens over b*L = 32768 rows
  qkv_proj_kernel<<<2048, 128, 0, stream>>>(x, W, qb, kb, vtb);
  // 16 batches x 32 blocks; each block = 4 waves x 16 query rows
  flash_attn_kernel<<<NBATCH * 32, 128, 0, stream>>>(qb, kb, vtb, out);
}
